// HybridGraphGPSBlock_47940424958279
// MI455X (gfx1250) — matrix operand order repaired and checked by reading, not compiled, not run on hardware
//
#include <hip/hip_runtime.h>

// ---------------------------------------------------------------------------
// MI455X (gfx1250) implementation of HybridGraphGPSBlock forward.
// Heavy 3x3 convs run as implicit GEMM on v_wmma_f32_16x16x32_bf16 (bf16 A/B,
// fp32 accumulate); InstanceNorm after every conv bounds bf16 rounding error.
// Weight K-chunks are streamed with global_load_async_to_lds_b32 (ASYNCcnt),
// A/B fragments are pre-staged in LDS in WMMA register layout so the inner
// loop is just ds_load_b128 pairs + v_wmma.
// ---------------------------------------------------------------------------

typedef __attribute__((ext_vector_type(16))) __bf16 v16bf;
typedef __attribute__((ext_vector_type(4)))  __bf16 v4bf;
typedef __attribute__((ext_vector_type(8)))  float  v8f;

__device__ __forceinline__ __bf16 f2bf(float f) {
  unsigned u = __builtin_bit_cast(unsigned, f);
  unsigned short h = (unsigned short)((u + 0x7FFFu + ((u >> 16) & 1u)) >> 16);
  return __builtin_bit_cast(__bf16, h);
}

// generic (flat) shared pointer -> 32-bit LDS byte offset (flat LDS aperture
// keeps the wave-relative LDS address in addr[31:0] per CDNA5 ISA 10.2)
__device__ __forceinline__ unsigned lds_off(const void* p) {
  return (unsigned)(unsigned long long)(__SIZE_TYPE__)p;
}

// ---------------------------------------------------------------------------
// concat-buffer init: h0[:, 0:16] = x, h0[:, 16:48] = 0
// ---------------------------------------------------------------------------
__global__ __launch_bounds__(256) void init_h0(const float* __restrict__ x,
                                               float* __restrict__ h0) {
  size_t idx = (size_t)blockIdx.x * 256 + threadIdx.x;
  size_t total = (size_t)2048 * 48 * 256;
  if (idx >= total) return;
  int s = (int)(idx & 255);
  int c = (int)((idx >> 8) % 48);
  int n = (int)(idx / (48 * 256));
  h0[idx] = (c < 16) ? x[((size_t)n * 16 + c) * 256 + s] : 0.f;
}

// ---------------------------------------------------------------------------
// symmetric signed edge pooling into channels [16:32) (pos) / [32:48) (neg)
// ---------------------------------------------------------------------------
__global__ __launch_bounds__(256) void edge_pool(const int* __restrict__ edges,
                                                 const float* __restrict__ x,
                                                 float* __restrict__ h0) {
  int e = blockIdx.x;
  int src = edges[e * 3 + 0];
  int sgn = edges[e * 3 + 1];
  int dst = edges[e * 3 + 2];
  int cblk = (sgn > 0) ? 16 : 32;
  for (int idx = threadIdx.x; idx < 16 * 256; idx += 256) {
    int c = idx >> 8, s = idx & 255;
    float vs = x[((size_t)src * 16 + c) * 256 + s];
    float vd = x[((size_t)dst * 16 + c) * 256 + s];
    atomicAdd(&h0[((size_t)dst * 48 + cblk + c) * 256 + s], vs);
    atomicAdd(&h0[((size_t)src * 48 + cblk + c) * 256 + s], vd);
  }
}

// ---------------------------------------------------------------------------
// Fused conv3x3(SAME) + bias + InstanceNorm + optional residual + LeakyReLU.
// One image per workgroup, 8 waves. Implicit GEMM:
//   M = 256 (spatial), N = COUT, K = CIN*9 (zero-padded to multiple of 32)
// ---------------------------------------------------------------------------
template <int CIN, int COUT>
__global__ __launch_bounds__(256) void conv3x3_in_lrelu(
    const float* __restrict__ in,   // [N][CIN][256]
    const float* __restrict__ wgt,  // [COUT][CIN][9]
    const float* __restrict__ bias, // [COUT]
    const float* __restrict__ gam,  // [COUT]
    const float* __restrict__ bet,  // [COUT]
    const float* __restrict__ res,  // [N][COUT][256] or nullptr
    float* __restrict__ out)        // [N][COUT][256]
{
  constexpr int K   = CIN * 9;
  constexpr int KP  = ((K + 31) / 32) * 32;
  constexpr int NT  = COUT / 16;
  constexpr int TPW = (16 * NT) / 8;   // tiles per wave (8 waves)

  __shared__ alignas(16) __bf16 Xs[CIN * 256];        // input image, bf16
  __shared__ alignas(32) __bf16 As[16 * 32 * 16];     // A frags for one k-chunk
  __shared__ alignas(32) __bf16 Bs[NT * 32 * 16];     // B frags for one k-chunk
  __shared__ alignas(16) float  Wgf[32 * COUT];       // async-staged fp32 weights
  __shared__ float ssum[COUT], ssq[COUT], scl[COUT], sft[COUT];

  const int n   = blockIdx.x;
  const int tid = threadIdx.x;
  const size_t ibase = (size_t)n * CIN * 256;
  const size_t obase = (size_t)n * COUT * 256;

  if (res) __builtin_prefetch(res + obase + tid * 4, 0, 1);

  // stage input image as bf16 (vectorized: b128 load, b64 LDS store)
  for (int idx = tid * 4; idx < CIN * 256; idx += 1024) {
    float4 v = *(const float4*)&in[ibase + idx];
    v4bf w4;
    w4[0] = f2bf(v.x); w4[1] = f2bf(v.y); w4[2] = f2bf(v.z); w4[3] = f2bf(v.w);
    *(v4bf*)&Xs[idx] = w4;
  }
  if (tid < COUT) { ssum[tid] = 0.f; ssq[tid] = 0.f; }
  __syncthreads();

  const int wv = tid >> 5, lane = tid & 31;
  const int kgS = lane >> 4, lnS = lane & 15;
  const __bf16 bz = __builtin_bit_cast(__bf16, (unsigned short)0);

  v8f acc[TPW];
#pragma unroll
  for (int i = 0; i < TPW; ++i)
    acc[i] = (v8f){0.f, 0.f, 0.f, 0.f, 0.f, 0.f, 0.f, 0.f};

  const unsigned long long wbase = (unsigned long long)(__SIZE_TYPE__)wgt;

  for (int kc = 0; kc < KP; kc += 32) {
    // --- async-stream this K-chunk of weights into LDS (fp32, [co][32]) ----
    for (int idx = tid; idx < 32 * COUT; idx += 256) {
      int co = idx >> 5, t = idx & 31;
      int kk = kc + t;
      if (kk < K) {
        unsigned lo = lds_off(&Wgf[idx]);
        unsigned go = (unsigned)((co * K + kk) * 4);
        asm volatile("global_load_async_to_lds_b32 %0, %1, %2"
                     :: "v"(lo), "v"(go), "s"(wbase) : "memory");
      }
    }

    // --- meanwhile: build A fragments for this chunk in WMMA reg layout ----
    // row = mt*32 + lane ; each thread owns rows tid and tid+256
#pragma unroll
    for (int half = 0; half < 2; ++half) {
      int mtS = (tid >> 5) + 8 * half;   // y block
      v16bf fr;
#pragma unroll
      for (int j = 0; j < 16; ++j) {
        // CDNA5 16-bit A layout: lanes 0-15 K={0..7,16..23}, lanes 16-31 +8
        int vv = j >> 1, hh = j & 1;
        int ko = (vv < 4) ? (kgS * 8 + vv * 2 + hh)
                          : (16 + kgS * 8 + (vv - 4) * 2 + hh);
        int k = kc + ko;
        __bf16 av = bz;
        if (k < K) {
          int ci  = k / 9;
          int tap = k - ci * 9;
          int yy  = mtS + tap / 3 - 1;
          int xx  = lnS + tap % 3 - 1;
          if (yy >= 0 && yy < 16 && xx >= 0 && xx < 16)
            av = Xs[ci * 256 + yy * 16 + xx];
        }
        fr[j] = av;
      }
      *(v16bf*)&As[(mtS * 32 + lane) * 16] = fr;
    }

    asm volatile("s_wait_asynccnt 0" ::: "memory");
    __syncthreads();   // Wgf + As visible to all waves

    // --- convert weight chunk into B fragments (bf16, WMMA reg layout) -----
    if (tid < NT * 32) {
      int nt = tid >> 5, lb = tid & 31;
      int kgB = lb >> 4, lnB = lb & 15;
      int co = nt * 16 + lnB;
      v16bf fr;
#pragma unroll
      for (int j = 0; j < 16; ++j) {
        int ko = (j < 8) ? (kgB * 8 + j) : (16 + kgB * 8 + (j - 8));
        int k = kc + ko;
        fr[j] = (k < K) ? f2bf(Wgf[co * 32 + ko]) : bz;
      }
      *(v16bf*)&Bs[(nt * 32 + lb) * 16] = fr;
    }
    __syncthreads();   // Bs ready

    // --- matrix core: two b128 LDS loads + one wmma per tile ---------------
#pragma unroll
    for (int i = 0; i < TPW; ++i) {
      int tile = wv + 8 * i;
      int mt = tile & 15;
      int nt = tile >> 4;
      v16bf a = *(const v16bf*)&As[(mt * 32 + lane) * 16];
      v16bf b = *(const v16bf*)&Bs[(nt * 32 + lane) * 16];
      acc[i] = __builtin_amdgcn_wmma_f32_16x16x32_bf16(
          false, a, false, b, (short)0, acc[i], false, false);
    }
    __syncthreads();   // As/Bs/Wgf free for next chunk
  }

  // bias + InstanceNorm statistics (exact fp32 via LDS atomics)
#pragma unroll
  for (int i = 0; i < TPW; ++i) {
    int nt = (wv + 8 * i) >> 4;
    int co = nt * 16 + lnS;
    float b0 = bias[co];
#pragma unroll
    for (int r = 0; r < 8; ++r) {
      float v = acc[i][r] + b0;
      acc[i][r] = v;
      atomicAdd(&ssum[co], v);
      atomicAdd(&ssq[co], v * v);
    }
  }
  __syncthreads();

  if (tid < COUT) {
    float mu  = ssum[tid] * (1.f / 256.f);
    float var = ssq[tid] * (1.f / 256.f) - mu * mu;
    float s   = gam[tid] * rsqrtf(var + 1e-5f);
    scl[tid]  = s;
    sft[tid]  = bet[tid] - mu * s;
  }
  __syncthreads();

  // normalize (+residual) + LeakyReLU; D-layout gives 8 contiguous m per lane
#pragma unroll
  for (int i = 0; i < TPW; ++i) {
    int tile = wv + 8 * i;
    int mt = tile & 15;
    int nt = tile >> 4;
    int co = nt * 16 + lnS;
    float s = scl[co], f = sft[co];
    size_t base = obase + (size_t)co * 256 + mt * 16 + 8 * kgS;
#pragma unroll
    for (int r = 0; r < 8; ++r) {
      float v = acc[i][r] * s + f;
      if (res) v += res[base + r];
      out[base + r] = (v > 0.f) ? v : 0.1f * v;
    }
  }
}

// ---------------------------------------------------------------------------
// pooled[n][c] = mean over spatial
// ---------------------------------------------------------------------------
__global__ __launch_bounds__(256) void pool_mean(const float* __restrict__ x,
                                                 float* __restrict__ pooled) {
  int t = blockIdx.x * 256 + threadIdx.x;
  if (t >= 2048 * 16) return;
  const float* p = x + (size_t)t * 256;
  float s = 0.f;
  for (int i = 0; i < 256; ++i) s += p[i];
  pooled[t] = s * (1.f / 256.f);
}

// ---------------------------------------------------------------------------
// Per-graph LN1 -> MHA (NH=4, hd=4) -> +pooled -> LN2 -> FFN(gelu) -> gdelta.
// Graphs are contiguous 32-node chains, so the argsort order is identity.
// One wave per graph; one thread per token.
// ---------------------------------------------------------------------------
__global__ __launch_bounds__(32) void attn_ffn(
    const float* __restrict__ pooled,
    const float* __restrict__ ln1g, const float* __restrict__ ln1b,
    const float* __restrict__ wqkv, const float* __restrict__ bqkv,
    const float* __restrict__ wo,   const float* __restrict__ bo,
    const float* __restrict__ ln2g, const float* __restrict__ ln2b,
    const float* __restrict__ w1,   const float* __restrict__ b1,
    const float* __restrict__ w2,   const float* __restrict__ b2,
    float* __restrict__ gdelta) {
  __shared__ float Ks[32][16], Vs[32][16];
  const int g = blockIdx.x, t = threadIdx.x;
  const int node = g * 32 + t;

  float p[16], tn[16];
  float mu = 0.f;
  for (int c = 0; c < 16; ++c) { p[c] = pooled[node * 16 + c]; mu += p[c]; }
  mu *= (1.f / 16.f);
  float var = 0.f;
  for (int c = 0; c < 16; ++c) { float d = p[c] - mu; var += d * d; }
  var *= (1.f / 16.f);
  float inv = rsqrtf(var + 1e-5f);
  for (int c = 0; c < 16; ++c) tn[c] = (p[c] - mu) * inv * ln1g[c] + ln1b[c];

  float q[16];
  for (int j = 0; j < 48; ++j) {
    float s = bqkv[j];
    for (int c = 0; c < 16; ++c) s += wqkv[j * 16 + c] * tn[c];
    if (j < 16) q[j] = s;
    else if (j < 32) Ks[t][j - 16] = s;
    else Vs[t][j - 32] = s;
  }
  __syncthreads();

  float o[16];
  for (int h = 0; h < 4; ++h) {
    float sc[32], m = -1e30f;
    for (int j = 0; j < 32; ++j) {
      float s = 0.f;
      for (int d = 0; d < 4; ++d) s += q[h * 4 + d] * Ks[j][h * 4 + d];
      s *= 0.5f;  // 1/sqrt(hd), hd=4
      sc[j] = s;
      m = fmaxf(m, s);
    }
    float den = 0.f;
    for (int j = 0; j < 32; ++j) { sc[j] = expf(sc[j] - m); den += sc[j]; }
    float r = 1.f / den;
    for (int d = 0; d < 4; ++d) {
      float s = 0.f;
      for (int j = 0; j < 32; ++j) s += sc[j] * Vs[j][h * 4 + d];
      o[h * 4 + d] = s * r;
    }
  }

  float tok[16];
  for (int c = 0; c < 16; ++c) {
    float s = bo[c];
    for (int d = 0; d < 16; ++d) s += wo[c * 16 + d] * o[d];
    tok[c] = p[c] + s;
  }

  // FFN with exact gelu
  mu = 0.f;
  for (int c = 0; c < 16; ++c) mu += tok[c];
  mu *= (1.f / 16.f);
  var = 0.f;
  for (int c = 0; c < 16; ++c) { float d = tok[c] - mu; var += d * d; }
  var *= (1.f / 16.f);
  inv = rsqrtf(var + 1e-5f);
  float t2[16];
  for (int c = 0; c < 16; ++c) t2[c] = (tok[c] - mu) * inv * ln2g[c] + ln2b[c];

  float hb[64];
  for (int j = 0; j < 64; ++j) {
    float s = b1[j];
    for (int c = 0; c < 16; ++c) s += w1[j * 16 + c] * t2[c];
    hb[j] = 0.5f * s * (1.f + erff(s * 0.7071067811865476f));
  }
  for (int c = 0; c < 16; ++c) {
    float s = b2[c];
    for (int j = 0; j < 64; ++j) s += w2[c * 64 + j] * hb[j];
    gdelta[node * 16 + c] = (tok[c] + s) - p[c];
  }
}

// ---------------------------------------------------------------------------
// xmid = x + local_scale*local_delta + global_scale*global_delta[n][c]
// ---------------------------------------------------------------------------
__global__ __launch_bounds__(256) void combine(const float* __restrict__ x,
                                               const float* __restrict__ ld,
                                               const float* __restrict__ gd,
                                               const float* __restrict__ ls,
                                               const float* __restrict__ gs,
                                               float* __restrict__ xmid) {
  size_t idx = (size_t)blockIdx.x * 256 + threadIdx.x;
  size_t total = (size_t)2048 * 16 * 256;
  if (idx >= total) return;
  size_t nc = idx >> 8;  // n*16 + c
  xmid[idx] = x[idx] + ls[0] * ld[idx] + gs[0] * gd[nc];
}

__global__ __launch_bounds__(256) void add2(const float* __restrict__ a,
                                            const float* __restrict__ b,
                                            float* __restrict__ o) {
  size_t idx = (size_t)blockIdx.x * 256 + threadIdx.x;
  size_t total = (size_t)2048 * 16 * 256;
  if (idx >= total) return;
  o[idx] = a[idx] + b[idx];
}

// ---------------------------------------------------------------------------
// host side
// ---------------------------------------------------------------------------
extern "C" void kernel_launch(void* const* d_in, const int* in_sizes, int n_in,
                              void* d_out, int out_size, void* d_ws,
                              size_t ws_size, hipStream_t stream) {
  // input order: x, edges, then params tree leaves (dict keys sorted, jax
  // pytree convention), then num_graphs, graph_size.
  const float* x     = (const float*)d_in[0];
  const int*   edges = (const int*)d_in[1];
  const float* a_bo    = (const float*)d_in[2];
  const float* a_bqkv  = (const float*)d_in[3];
  const float* a_wo    = (const float*)d_in[4];
  const float* a_wqkv  = (const float*)d_in[5];
  const float* c1_b    = (const float*)d_in[6];
  const float* c1_w    = (const float*)d_in[7];
  const float* c2_b    = (const float*)d_in[8];
  const float* c2_w    = (const float*)d_in[9];
  const float* n1_b    = (const float*)d_in[10];
  const float* n1_g    = (const float*)d_in[11];
  const float* n2_b    = (const float*)d_in[12];
  const float* n2_g    = (const float*)d_in[13];
  const float* r1c1_b  = (const float*)d_in[14];
  const float* r1c1_w  = (const float*)d_in[15];
  const float* r1c2_b  = (const float*)d_in[16];
  const float* r1c2_w  = (const float*)d_in[17];
  const float* r1n1_b  = (const float*)d_in[18];
  const float* r1n1_g  = (const float*)d_in[19];
  const float* r1n2_b  = (const float*)d_in[20];
  const float* r1n2_g  = (const float*)d_in[21];
  const float* r2c1_b  = (const float*)d_in[22];
  const float* r2c1_w  = (const float*)d_in[23];
  const float* r2c2_b  = (const float*)d_in[24];
  const float* r2c2_w  = (const float*)d_in[25];
  const float* r2n1_b  = (const float*)d_in[26];
  const float* r2n1_g  = (const float*)d_in[27];
  const float* r2n2_b  = (const float*)d_in[28];
  const float* r2n2_g  = (const float*)d_in[29];
  const float* f_b1    = (const float*)d_in[30];
  const float* f_b2    = (const float*)d_in[31];
  const float* f_w1    = (const float*)d_in[32];
  const float* f_w2    = (const float*)d_in[33];
  const float* gscale  = (const float*)d_in[34];
  const float* ln1_b   = (const float*)d_in[35];
  const float* ln1_g   = (const float*)d_in[36];
  const float* ln2_b   = (const float*)d_in[37];
  const float* ln2_g   = (const float*)d_in[38];
  const float* lscale  = (const float*)d_in[39];
  const float* rc_b    = (const float*)d_in[40];
  const float* rc_w    = (const float*)d_in[41];
  const float* rn_b    = (const float*)d_in[42];
  const float* rn_g    = (const float*)d_in[43];
  const float* rbc1_b  = (const float*)d_in[44];
  const float* rbc1_w  = (const float*)d_in[45];
  const float* rbc2_b  = (const float*)d_in[46];
  const float* rbc2_w  = (const float*)d_in[47];
  const float* rbn1_b  = (const float*)d_in[48];
  const float* rbn1_g  = (const float*)d_in[49];
  const float* rbn2_b  = (const float*)d_in[50];
  const float* rbn2_g  = (const float*)d_in[51];

  const int nE = in_sizes[1] / 3;

  const size_t SZ48 = (size_t)2048 * 48 * 256;
  const size_t SZ16 = (size_t)2048 * 16 * 256;
  float* bufA   = (float*)d_ws;          // concat / ping-pong
  float* bufB   = bufA + SZ48;
  float* bufC   = bufB + SZ48;
  float* ldel   = bufC + SZ48;           // local_delta (16 ch)
  float* xmid   = ldel + SZ16;
  float* rA     = xmid + SZ16;
  float* rB     = rA + SZ16;
  float* pooled = rB + SZ16;             // [2048][16]
  float* gdelta = pooled + 2048 * 16;    // [2048][16]

  // 1) build concat [x | pos | neg]
  init_h0<<<(int)(SZ48 / 256), 256, 0, stream>>>(x, bufA);
  edge_pool<<<nE, 256, 0, stream>>>(edges, x, bufA);

  // 2) cmp trunk: conv48->48 (+IN+lrelu), two resblocks, conv48->16
  conv3x3_in_lrelu<48, 48><<<2048, 256, 0, stream>>>(bufA, c1_w, c1_b, n1_g, n1_b, nullptr, bufB);     // h1
  conv3x3_in_lrelu<48, 48><<<2048, 256, 0, stream>>>(bufB, r1c1_w, r1c1_b, r1n1_g, r1n1_b, nullptr, bufC);
  conv3x3_in_lrelu<48, 48><<<2048, 256, 0, stream>>>(bufC, r1c2_w, r1c2_b, r1n2_g, r1n2_b, bufB, bufA); // h3
  conv3x3_in_lrelu<48, 48><<<2048, 256, 0, stream>>>(bufA, r2c1_w, r2c1_b, r2n1_g, r2n1_b, nullptr, bufC);
  conv3x3_in_lrelu<48, 48><<<2048, 256, 0, stream>>>(bufC, r2c2_w, r2c2_b, r2n2_g, r2n2_b, bufA, bufB); // h5
  conv3x3_in_lrelu<48, 16><<<2048, 256, 0, stream>>>(bufB, c2_w, c2_b, n2_g, n2_b, nullptr, ldel);      // local_delta

  // 3) global branch: pooled means, per-graph attention+FFN
  pool_mean<<<(2048 * 16 + 255) / 256, 256, 0, stream>>>(x, pooled);
  attn_ffn<<<64, 32, 0, stream>>>(pooled, ln1_g, ln1_b, a_wqkv, a_bqkv, a_wo,
                                  a_bo, ln2_g, ln2_b, f_w1, f_b1, f_w2, f_b2,
                                  gdelta);

  // 4) x = x + ls*local_delta + gs*global_delta
  combine<<<(int)(SZ16 / 256), 256, 0, stream>>>(x, ldel, gdelta, lscale, gscale, xmid);

  // 5) refinement: resblock(16) then conv16->16, out = xmid + xr
  conv3x3_in_lrelu<16, 16><<<2048, 256, 0, stream>>>(xmid, rbc1_w, rbc1_b, rbn1_g, rbn1_b, nullptr, rA);
  conv3x3_in_lrelu<16, 16><<<2048, 256, 0, stream>>>(rA, rbc2_w, rbc2_b, rbn2_g, rbn2_b, xmid, rB);
  conv3x3_in_lrelu<16, 16><<<2048, 256, 0, stream>>>(rB, rc_w, rc_b, rn_g, rn_b, nullptr, rA);
  add2<<<(int)(SZ16 / 256), 256, 0, stream>>>(xmid, rA, (float*)d_out);

  (void)n_in; (void)out_size; (void)ws_size;
}
